// Codebook_9560597201278
// MI455X (gfx1250) — matrix operand order repaired and checked by reading, not compiled, not run on hardware
//
#include <hip/hip_runtime.h>
#include <hip/hip_bf16.h>
#include <math.h>

// ---------------------------------------------------------------------------
// VQ-VAE codebook quantization for MI455X (gfx1250, wave32, WMMA).
//
// x [16,1024,32,32] f32 (= [B=16][D=1024][P=1024], row n = b*P+p)
// cb [K=1024, D=1024] f32
// Outputs (flat f32): z_out (16777216) | loss (1) | perplexity (1) | idx (16384)
//
// Distance GEMM via v_wmma_f32_16x16x32_f16 (f32 accum). Register-blocked
// 4 m-tiles x 2 n-tiles per wave so each WMMA costs only 0.5KB LDS + 0.25KB L2.
// ---------------------------------------------------------------------------

typedef __attribute__((ext_vector_type(16))) _Float16 v16h;
typedef __attribute__((ext_vector_type(8)))  float    v8f;

#define NB   16
#define DDIM 1024
#define PDIM 1024
#define NROW 16384
#define KCB  1024
#define NTOT 16777216
#define OUT_LOSS 16777216
#define OUT_PERP 16777217
#define OUT_IDX  16777218

// ---------------- ws layout (bytes) ----------------
#define WS_BPACK 0            // f16 [64 nt][32 dt][32 lane][16] = 2 MB
#define WS_ENORM 2097152      // f32 [1024]
#define WS_IDX   2101248      // int [16384]
#define WS_HIST  2166784      // u32 [1024]
#define WS_SSE   2170880      // f32 [1]

// ---------------------------------------------------------------------------
__global__ void vq_zero(unsigned* __restrict__ hist, float* __restrict__ sse) {
  int t = threadIdx.x;
  if (t < KCB) hist[t] = 0u;
  if (t == 0)  *sse = 0.0f;
}

// Pack codebook into WMMA B-fragment order (32x16 f16 per fragment):
// lane&15 = n(code), (lane>>4)*16 = K offset, 16 consecutive K per lane.
__global__ void vq_pack_b(const float* __restrict__ cb, v16h* __restrict__ Bp) {
  int t    = threadIdx.x;
  int frag = blockIdx.x * 8 + (t >> 5);     // 2048 fragments total
  int lane = t & 31;
  int nt = frag >> 5, dt = frag & 31;
  int kc   = nt * 16 + (lane & 15);
  int koff = (lane >> 4) * 16;
  const float* src = cb + (size_t)kc * DDIM + dt * 32 + koff;
  v16h v;
#pragma unroll
  for (int h = 0; h < 16; ++h) v[h] = (_Float16)src[h];
  Bp[frag * 32 + lane] = v;                 // contiguous 32B per lane
}

__global__ void vq_norms(const float* __restrict__ cb, float* __restrict__ en) {
  __shared__ float red[256];
  int k = blockIdx.x, t = threadIdx.x;
  float s = 0.0f;
  for (int i = t; i < DDIM; i += 256) { float v = cb[(size_t)k * DDIM + i]; s += v * v; }
  red[t] = s; __syncthreads();
  for (int st = 128; st; st >>= 1) { if (t < st) red[t] += red[t + st]; __syncthreads(); }
  if (t == 0) en[k] = red[0];
}

// ---------------------------------------------------------------------------
// GEMM + running argmin.  256 WGs x 256 threads (8 waves).
// WG owns 4 m-tiles (A staged once into 128KB LDS, shared by all 8 waves).
// Wave w handles n-tiles {g*16 + 2w, g*16 + 2w + 1} for g = 0..3.
// Per dt: 4 A fragments (LDS) + 2 B fragments (L2) -> 8 WMMAs.
// ---------------------------------------------------------------------------
__global__ void vq_gemm_argmin(const float* __restrict__ x,
                               const v16h* __restrict__ Bp,
                               const float* __restrict__ enorm,
                               int* __restrict__ idx_out,
                               unsigned* __restrict__ hist,
                               float* __restrict__ out_idx_f) {
  extern __shared__ char smem[];
  _Float16* ldsA  = (_Float16*)smem;                     // 4 * 16384 halves = 128 KB
  float*    candD = (float*)(smem + 4 * 32768);          // [4][16][8]
  int*      candI = (int*)(smem + 4 * 32768 + 2048);     // [4][16][8]

  const int tid  = threadIdx.x;
  const int w    = tid >> 5, lane = tid & 31;
  const int lh   = lane & 15, lsel = lane >> 4;

  // ---- stage A: wave w stages m-tile (w&3), dt in [(w>>2)*16, +16) ----
  {
    const int mloc = w & 3;
    const int mt   = blockIdx.x * 4 + mloc;
    const int m    = mt * 16 + lh;            // 16 consecutive rows share b
    const int b    = m >> 10, p = m & 1023;
    const int koff8 = lsel * 8;
    const float* xb = x + (size_t)b * (DDIM * PDIM) + p;
    for (int dt = (w >> 2) * 16; dt < (w >> 2) * 16 + 16; ++dt) {
      v16h av;
#pragma unroll
      for (int h = 0; h < 16; ++h) {
        int kk = 2 * koff8 + h;  // A frag K order
        av[h] = (_Float16)xb[(size_t)(dt * 32 + kk) * PDIM];
      }
      *(v16h*)(ldsA + (((size_t)mloc * 32 + dt) * 32 + lane) * 16) = av;
    }
  }
  __syncthreads();

  float bestD[4][8]; int bestI[4][8];
#pragma unroll
  for (int mi = 0; mi < 4; ++mi)
#pragma unroll
    for (int j = 0; j < 8; ++j) { bestD[mi][j] = 3.0e38f; bestI[mi][j] = 0; }

  for (int g = 0; g < 4; ++g) {
    const int nt0 = g * 16 + w * 2;
    v8f acc[4][2];
#pragma unroll
    for (int mi = 0; mi < 4; ++mi) { acc[mi][0] = (v8f){}; acc[mi][1] = (v8f){}; }

    const v16h* bp = Bp + (size_t)nt0 * 1024 + lane;     // frag stride = 32 v16h
    __builtin_prefetch(bp + 2048, 0, 0);                 // n-tile nt0+2's B stream
#pragma unroll 2
    for (int dt = 0; dt < 32; ++dt) {
      v16h b0 = bp[dt * 32];                             // n-tile nt0
      v16h b1 = bp[1024 + dt * 32];                      // n-tile nt0+1
#pragma unroll
      for (int mi = 0; mi < 4; ++mi) {
        v16h a = *(const v16h*)(ldsA + (((size_t)mi * 32 + dt) * 32 + lane) * 16);
        acc[mi][0] = __builtin_amdgcn_wmma_f32_16x16x32_f16(false, a, false, b0,
                                                            (short)0, acc[mi][0],
                                                            false, false);
        acc[mi][1] = __builtin_amdgcn_wmma_f32_16x16x32_f16(false, a, false, b1,
                                                            (short)0, acc[mi][1],
                                                            false, false);
      }
    }
    const int   kc0 = nt0 * 16 + lh, kc1 = kc0 + 16;
    const float e0  = enorm[kc0],    e1  = enorm[kc1];
#pragma unroll
    for (int mi = 0; mi < 4; ++mi)
#pragma unroll
      for (int j = 0; j < 8; ++j) {
        float d0 = e0 - 2.0f * acc[mi][0][j];   // ||e||^2 - 2 x.e
        float d1 = e1 - 2.0f * acc[mi][1][j];
        if (d0 < bestD[mi][j]) { bestD[mi][j] = d0; bestI[mi][j] = kc0; }
        if (d1 < bestD[mi][j]) { bestD[mi][j] = d1; bestI[mi][j] = kc1; }
      }
  }

  // cross-lane argmin within each 16-lane group (tie -> smaller index)
#pragma unroll
  for (int mi = 0; mi < 4; ++mi)
#pragma unroll
    for (int j = 0; j < 8; ++j) {
      float dv = bestD[mi][j]; int iv = bestI[mi][j];
      for (int off = 8; off; off >>= 1) {
        float od = __shfl_xor(dv, off, 32);
        int   oi = __shfl_xor(iv, off, 32);
        if (od < dv || (od == dv && oi < iv)) { dv = od; iv = oi; }
      }
      if (lh == 0) {
        int rloc = j + 8 * lsel;
        candD[((mi * 16) + rloc) * 8 + w] = dv;
        candI[((mi * 16) + rloc) * 8 + w] = iv;
      }
    }
  __syncthreads();

  // 8-way merge across waves; lexicographic (dist, idx) keeps smallest index
  if (tid < 64) {
    int mi = tid >> 4, rloc = tid & 15;
    float bd = candD[((mi * 16) + rloc) * 8 + 0];
    int   bi = candI[((mi * 16) + rloc) * 8 + 0];
#pragma unroll
    for (int ww = 1; ww < 8; ++ww) {
      float d = candD[((mi * 16) + rloc) * 8 + ww];
      int   i = candI[((mi * 16) + rloc) * 8 + ww];
      if (d < bd || (d == bd && i < bi)) { bd = d; bi = i; }
    }
    int row = (blockIdx.x * 4 + mi) * 16 + rloc;
    idx_out[row]   = bi;
    out_idx_f[row] = (float)bi;
    atomicAdd(&hist[bi], 1u);
  }
}

// ---------------------------------------------------------------------------
// Gather z_q = cb[idx], write z_out in [B,D,P] layout, accumulate SSE (f32).
// LDS transpose tile (pitch 129 -> conflict-free banks).
// ---------------------------------------------------------------------------
__global__ void vq_gather(const float* __restrict__ x, const float* __restrict__ cb,
                          const int* __restrict__ idx, float* __restrict__ zout,
                          float* __restrict__ sse) {
  __shared__ float tile[32][129];
  __shared__ float red[256];
  __shared__ int   rowidx[32];
  const int bB = blockIdx.z;
  const int p0 = blockIdx.x * 32;
  const int d0 = blockIdx.y * 128;
  const int t  = threadIdx.x;

  if (t < 32) rowidx[t] = idx[bB * PDIM + p0 + t];
  __syncthreads();
  for (int i = t; i < 32 * 128; i += 256) {
    int pl = i >> 7, dl = i & 127;                       // coalesced over d
    tile[pl][dl] = cb[(size_t)rowidx[pl] * DDIM + d0 + dl];
  }
  __syncthreads();
  float acc = 0.0f;
  for (int i = t; i < 32 * 128; i += 256) {
    int dl = i >> 5, pl = i & 31;                        // coalesced over p
    float zv = tile[pl][dl];
    size_t off = ((size_t)bB * DDIM + d0 + dl) * PDIM + p0 + pl;
    float df = zv - x[off];
    acc += df * df;
    zout[off] = zv;
  }
  red[t] = acc; __syncthreads();
  for (int s = 128; s; s >>= 1) { if (t < s) red[t] += red[t + s]; __syncthreads(); }
  if (t == 0) atomicAdd(sse, red[0]);
}

__global__ void vq_finalize(const unsigned* __restrict__ hist,
                            const float* __restrict__ sse,
                            float* __restrict__ out) {
  __shared__ float red[256];
  int t = threadIdx.x;
  float s = 0.0f;
  for (int k = t; k < KCB; k += 256) {
    float c = (float)hist[k] * (1.0f / (float)NROW);
    s += c * logf(c + 1e-10f);
  }
  red[t] = s; __syncthreads();
  for (int st = 128; st; st >>= 1) { if (t < st) red[t] += red[t + st]; __syncthreads(); }
  if (t == 0) {
    out[OUT_LOSS] = (*sse) * 1.25f / (float)NTOT;  // codebook + BETA*commitment
    out[OUT_PERP] = expf(-red[0]);
  }
}

// ---------------------------------------------------------------------------
extern "C" void kernel_launch(void* const* d_in, const int* in_sizes, int n_in,
                              void* d_out, int out_size, void* d_ws, size_t ws_size,
                              hipStream_t stream) {
  (void)in_sizes; (void)n_in; (void)out_size; (void)ws_size;
  const float* x  = (const float*)d_in[0];
  const float* cb = (const float*)d_in[1];
  char* ws = (char*)d_ws;
  v16h*     Bp    = (v16h*)(ws + WS_BPACK);
  float*    enorm = (float*)(ws + WS_ENORM);
  int*      idxb  = (int*)(ws + WS_IDX);
  unsigned* hist  = (unsigned*)(ws + WS_HIST);
  float*    sse   = (float*)(ws + WS_SSE);
  float*    out   = (float*)d_out;

  vq_zero   <<<1, 1024, 0, stream>>>(hist, sse);
  vq_pack_b <<<256, 256, 0, stream>>>(cb, Bp);
  vq_norms  <<<1024, 256, 0, stream>>>(cb, enorm);
  vq_gemm_argmin<<<256, 256, 4 * 32768 + 4096, stream>>>(x, Bp, enorm, idxb, hist,
                                                         out + OUT_IDX);
  vq_gather <<<dim3(32, 8, 16), 256, 0, stream>>>(x, cb, idxb, out, sse);
  vq_finalize<<<1, 256, 0, stream>>>(hist, sse, out);
}
